// MultiHeadAttention_42588895707911
// MI455X (gfx1250) — matrix operand order repaired and checked by reading, not compiled, not run on hardware
//
#include <hip/hip_runtime.h>

// MI455X (gfx1250) multi-head cross-attention, flash-style, all GEMMs on
// v_wmma_f32_16x16x32_f16. All WMMA operands are pre-swizzled in memory into
// a contiguous-k lane layout (k = 16*(lane/16) + i) so hot-loop operand fetch is one
// contiguous 32B-per-lane load (global_load_b128 x2 or ds_load_b128 x2).
//
// Pipeline:
//   0) repack_a : x, memory (f32) -> A-layout f16 tiles
//      repack_b : wq,wk,wv,wo (f32) -> B-layout f16 tiles
//   1) qkv_proj : 32x64 tile/wave (8 WMMA per k-chunk).
//                 Q -> A-layout tiles (pre-scaled 1/sqrt(DH)),
//                 K -> [B,H,S,DH] f16 (B^T operand contiguous there),
//                 V -> B-layout tiles
//   2) attn     : 4 q-tile waves per block share K/V chunks staged in LDS;
//                 streaming softmax; output written as A-layout f16 tiles
//   3) out_proj : 32x64 tile/wave, out = attn @ wo + bo -> f32 d_out

typedef __attribute__((ext_vector_type(16))) _Float16 v16h;
typedef __attribute__((ext_vector_type(8)))  _Float16 v8h;
typedef __attribute__((ext_vector_type(8)))  float    v8f;

constexpr int Bc  = 2;
constexpr int Sc  = 2048;
constexpr int Dc  = 1024;
constexpr int Hc  = 16;
constexpr int DHc = 64;
constexpr int Mc  = Bc * Sc;          // 4096 GEMM rows

constexpr size_t TILE = 512;          // one WMMA operand: 32 lanes x 16 halfs

// ---------------------------------------------------------------------------
// Layout maps used here (contiguous k order, wave32):
// A (16x32 f16):  lane = (row&15) + (k&16 ? 16 : 0)
//                 i    = k&15
// B (32x16 f16):  lane = (n&15) + (k&16 ? 16 : 0) ;  i = k&15
// C/D (16x16 f32): lane = col + 16*(row>=8) ; vgpr r = row&7
// ---------------------------------------------------------------------------

__device__ __forceinline__ int a_lane(int row, int k) {
    return (row & 15) + ((k & 16) ? 16 : 0);
}
__device__ __forceinline__ int a_idx(int k) {
    return k & 15;
}
__device__ __forceinline__ int a_k_of(int lane, int i) {
    return ((lane & 16) ? 16 : 0) + i;
}

__device__ __forceinline__ v8f wmma_f16(v16h a, v16h b, v8f c) {
    return __builtin_amdgcn_wmma_f32_16x16x32_f16(false, a, false, b,
                                                  (short)0, c, false, false);
}

// ---------------------------------------------------------------------------
// Repack kernels (one-time, memory-bound)
// ---------------------------------------------------------------------------

__global__ void __launch_bounds__(128)
repack_a_kernel(const float* __restrict__ src, _Float16* __restrict__ dst) {
    const int wave = threadIdx.x >> 5;
    const int kc   = blockIdx.x * 4 + wave;      // 0..31
    const int mt   = blockIdx.y;                 // 0..255
    const int lane = threadIdx.x & 31;
    const int row  = lane & 15;

    const float* s = src + (size_t)(mt * 16 + row) * Dc + kc * 32;
    v16h a;
#pragma unroll
    for (int i = 0; i < 16; ++i) a[i] = (_Float16)s[a_k_of(lane, i)];
    *(v16h*)(dst + ((size_t)mt * 32 + kc) * TILE + lane * 16) = a;
}

__global__ void __launch_bounds__(128)
repack_b_kernel(const float* __restrict__ w0, const float* __restrict__ w1,
                const float* __restrict__ w2, const float* __restrict__ w3,
                _Float16* __restrict__ dstBase) {
    const int z = blockIdx.z;
    const float* w = (z == 0) ? w0 : (z == 1) ? w1 : (z == 2) ? w2 : w3;
    _Float16* dst = dstBase + (size_t)z * (Dc / 32) * (Dc / 16) * TILE;

    const int wave = threadIdx.x >> 5;
    const int nt   = blockIdx.x * 4 + wave;      // 0..63
    const int kc   = blockIdx.y;                 // 0..31
    const int lane = threadIdx.x & 31;
    const int n    = lane & 15;
    const int kb   = (lane & 16) ? 16 : 0;

    v16h b;
#pragma unroll
    for (int i = 0; i < 16; ++i)
        b[i] = (_Float16)w[(size_t)(kc * 32 + kb + i) * Dc + nt * 16 + n];
    *(v16h*)(dst + ((size_t)kc * 64 + nt) * TILE + lane * 16) = b;
}

// ---------------------------------------------------------------------------
// Kernel 1: fused QKV projections. grid.z: 0=Q, 1=K, 2=V. 32x64 tile / wave.
// ---------------------------------------------------------------------------
__global__ void __launch_bounds__(128)
qkv_proj_kernel(const _Float16* __restrict__ xa, const _Float16* __restrict__ ma,
                const _Float16* __restrict__ wB,
                const float* __restrict__ bq, const float* __restrict__ bk,
                const float* __restrict__ bv,
                _Float16* __restrict__ Qa, _Float16* __restrict__ Kb16,
                _Float16* __restrict__ Vb) {
    const int z = blockIdx.z;
    const _Float16* A = (z == 0) ? xa : ma;
    const _Float16* W = wB + (size_t)z * (Dc / 32) * (Dc / 16) * TILE;
    const float* bias = (z == 0) ? bq : (z == 1) ? bk : bv;
    const float scale = (z == 0) ? 0.125f : 1.0f;   // 1/sqrt(64)

    const int wave = threadIdx.x >> 5;
    const int mtb  = (blockIdx.y * 4 + wave) * 2;   // pair of 16-row tiles
    const int nt0  = blockIdx.x * 4;
    const int lane = threadIdx.x & 31;

    v8f acc[2][4] = {};
#pragma unroll 2
    for (int kc = 0; kc < Dc / 32; ++kc) {
        const v16h a0 = *(const v16h*)(A + ((size_t)(mtb + 0) * 32 + kc) * TILE + lane * 16);
        const v16h a1 = *(const v16h*)(A + ((size_t)(mtb + 1) * 32 + kc) * TILE + lane * 16);
#pragma unroll
        for (int t = 0; t < 4; ++t) {
            const v16h b = *(const v16h*)(W + ((size_t)kc * 64 + nt0 + t) * TILE + lane * 16);
            acc[0][t] = wmma_f16(a0, b, acc[0][t]);
            acc[1][t] = wmma_f16(a1, b, acc[1][t]);
        }
    }

    const int rowoff = (lane & 16) ? 8 : 0;
    const int cl     = lane & 15;
#pragma unroll
    for (int t = 0; t < 4; ++t) {
        const int   col = (nt0 + t) * 16 + cl;
        const float bsv = bias[col];
        const int   h   = col >> 6;
        const int   dh  = col & 63;
#pragma unroll
        for (int mi = 0; mi < 2; ++mi)
#pragma unroll
        for (int r = 0; r < 8; ++r) {
            const int m  = (mtb + mi) * 16 + r + rowoff;
            const int bb = m >> 11;
            const int s  = m & 2047;
            const int head = bb * Hc + h;
            const _Float16 val = (_Float16)((acc[mi][t][r] + bsv) * scale);
            if (z == 0) {
                const int qt  = s >> 4;
                const int kc2 = dh >> 5;
                const int k   = dh & 31;
                Qa[(((size_t)head * (Sc / 16) + qt) * 2 + kc2) * TILE
                   + a_lane(s, k) * 16 + a_idx(k)] = val;
            } else if (z == 1) {
                Kb16[((size_t)head * Sc + s) * DHc + dh] = val;
            } else {
                const int kt = s >> 5, t2 = dh >> 4, nn = dh & 15, kl = s & 31;
                Vb[(((size_t)head * (Sc / 32) + kt) * 4 + t2) * TILE
                   + (nn + ((kl & 16) ? 16 : 0)) * 16 + (kl & 15)] = val;
            }
        }
    }
}

// ---------------------------------------------------------------------------
// Kernel 2: flash attention. 4 q-tile waves / block, shared LDS K/V staging.
// ---------------------------------------------------------------------------
__global__ void __launch_bounds__(128)
attn_kernel(const _Float16* __restrict__ Qa, const _Float16* __restrict__ Kb16,
            const _Float16* __restrict__ Vb, const unsigned char* __restrict__ mask,
            _Float16* __restrict__ attnA) {
    const int wave = threadIdx.x >> 5;
    const int qt   = blockIdx.x * 4 + wave;     // q tile (16 rows)
    const int h    = blockIdx.y;
    const int b    = blockIdx.z;
    const int head = b * Hc + h;
    const int tid  = threadIdx.x;

    const _Float16* K = Kb16 + (size_t)head * Sc * DHc;
    const _Float16* V = Vb + (size_t)head * (Sc / 32) * 4 * TILE;
    const unsigned char* mrow = mask + (size_t)b * Sc;

    __shared__ _Float16 ldsK[32 * 64];    // raw K chunk   (4KB)
    __shared__ _Float16 ldsV[4 * 512];    // V B-tiles     (4KB)
    __shared__ _Float16 ldsP[4 * 512];    // per-wave P    (4KB)
    _Float16* ldsPw = ldsP + wave * 512;

    const int lane  = threadIdx.x & 31;
    const int half8 = (lane & 16) ? 8 : 0;
    const int cl    = lane & 15;
    const int kb16  = (lane & 16) ? 16 : 0;

    const v16h aq0 = *(const v16h*)(Qa + (((size_t)head * (Sc / 16) + qt) * 2 + 0) * TILE + lane * 16);
    const v16h aq1 = *(const v16h*)(Qa + (((size_t)head * (Sc / 16) + qt) * 2 + 1) * TILE + lane * 16);

    constexpr float NEG = -3.0e30f;
    float m_r[8], l_r[8];
    v8f   O[4] = {};
#pragma unroll
    for (int r = 0; r < 8; ++r) { m_r[r] = NEG; l_r[r] = 0.0f; }

    for (int k0 = 0; k0 < Sc; k0 += 32) {
        // ---- cooperative stage of K chunk + V tiles (8KB, contiguous) ----
        const _Float16* Kg = K + (size_t)k0 * DHc;              // 32*64 halfs
        const _Float16* Vg = V + ((size_t)(k0 >> 5) * 4) * TILE; // 4*512 halfs
        const v8h kr0 = *(const v8h*)(Kg + tid * 8);
        const v8h kr1 = *(const v8h*)(Kg + (tid + 128) * 8);
        const v8h vr0 = *(const v8h*)(Vg + tid * 8);
        const v8h vr1 = *(const v8h*)(Vg + (tid + 128) * 8);
        if (k0 + 32 < Sc) {   // gfx1250 global_prefetch of next chunk
            __builtin_prefetch(Kg + 32 * DHc + tid * 16, 0, 1);
            __builtin_prefetch(Vg + 4 * TILE + tid * 16, 0, 1);
        }
        __syncthreads();                       // prior chunk fully consumed
        *(v8h*)(ldsK + tid * 8) = kr0;
        *(v8h*)(ldsK + (tid + 128) * 8) = kr1;
        *(v8h*)(ldsV + tid * 8) = vr0;
        *(v8h*)(ldsV + (tid + 128) * 8) = vr1;
        __syncthreads();                       // stage visible to all waves

        // ---- logits: QK^T from LDS (B^T operand contiguous per lane) ----
        v8f c0 = {}, c1 = {};
        {
            const v16h bk00 = *(const v16h*)(ldsK + (size_t)cl * DHc + 0  + kb16);
            const v16h bk01 = *(const v16h*)(ldsK + (size_t)cl * DHc + 32 + kb16);
            const v16h bk10 = *(const v16h*)(ldsK + (size_t)(16 + cl) * DHc + 0  + kb16);
            const v16h bk11 = *(const v16h*)(ldsK + (size_t)(16 + cl) * DHc + 32 + kb16);
            c0 = wmma_f16(aq0, bk00, c0);
            c0 = wmma_f16(aq1, bk01, c0);
            c1 = wmma_f16(aq0, bk10, c1);
            c1 = wmma_f16(aq1, bk11, c1);
        }

        // ---- mask (per-lane select, EXEC stays all-1s) ----
        const bool msk0 = mrow[k0 + cl]      != 0;
        const bool msk1 = mrow[k0 + 16 + cl] != 0;
#pragma unroll
        for (int r = 0; r < 8; ++r) {
            c0[r] = msk0 ? NEG : c0[r];
            c1[r] = msk1 ? NEG : c1[r];
        }

        // ---- online softmax ----
        float sc_r[8];
#pragma unroll
        for (int r = 0; r < 8; ++r) {
            float rm = fmaxf(c0[r], c1[r]);
            rm = fmaxf(rm, __shfl_xor(rm, 1, 32));
            rm = fmaxf(rm, __shfl_xor(rm, 2, 32));
            rm = fmaxf(rm, __shfl_xor(rm, 4, 32));
            rm = fmaxf(rm, __shfl_xor(rm, 8, 32));
            const float mn = fmaxf(m_r[r], rm);
            const float sc = __expf(m_r[r] - mn);
            m_r[r] = mn;
            const float p0 = __expf(c0[r] - mn);
            const float p1 = __expf(c1[r] - mn);
            c0[r] = p0; c1[r] = p1;
            float rs = p0 + p1;
            rs += __shfl_xor(rs, 1, 32);
            rs += __shfl_xor(rs, 2, 32);
            rs += __shfl_xor(rs, 4, 32);
            rs += __shfl_xor(rs, 8, 32);
            l_r[r] = l_r[r] * sc + rs;
            sc_r[r] = sc;
        }

#pragma unroll
        for (int t = 0; t < 4; ++t)
#pragma unroll
            for (int r = 0; r < 8; ++r) O[t][r] *= sc_r[r];

        // ---- P: C/D layout -> LDS (per-wave region) -> A layout ----
#pragma unroll
        for (int r = 0; r < 8; ++r) {
            ldsPw[(r + half8) * 32 + cl]      = (_Float16)c0[r];
            ldsPw[(r + half8) * 32 + 16 + cl] = (_Float16)c1[r];
        }
        asm volatile("s_wait_dscnt 0" ::: "memory");
        v16h ap;
        {
            const int row = lane & 15;
            const int kb  = (lane & 16) ? 16 : 0;
            const v8h lo = *(const v8h*)&ldsPw[row * 32 + kb];
            const v8h hi = *(const v8h*)&ldsPw[row * 32 + 8 + kb];
#pragma unroll
            for (int i = 0; i < 8; ++i) { ap[i] = lo[i]; ap[8 + i] = hi[i]; }
        }

        // ---- P x V from LDS (pre-swizzled tiles) ----
#pragma unroll
        for (int t = 0; t < 4; ++t)
            O[t] = wmma_f16(ap, *(const v16h*)(ldsV + t * TILE + lane * 16), O[t]);
    }

    // ---- normalize; store as A-layout f16 tiles for out-proj ----
    float inv_r[8];
#pragma unroll
    for (int r = 0; r < 8; ++r) inv_r[r] = 1.0f / l_r[r];

    const size_t mt = (size_t)(b * Sc + qt * 16) >> 4;
#pragma unroll
    for (int t = 0; t < 4; ++t) {
        const int col = h * DHc + t * 16 + cl;
        const int kc  = col >> 5;
        const int k   = col & 31;
        const int ln  = a_lane(0, k);
        const int ii  = a_idx(k);
#pragma unroll
        for (int r = 0; r < 8; ++r) {
            const int row15 = r + half8;
            attnA[((size_t)mt * 32 + kc) * TILE + (ln + row15) * 16 + ii] =
                (_Float16)(O[t][r] * inv_r[r]);
        }
    }
}

// ---------------------------------------------------------------------------
// Kernel 3: output projection, 32x64 tile / wave. out = attn @ wo + bo (f32)
// ---------------------------------------------------------------------------
__global__ void __launch_bounds__(128)
out_proj_kernel(const _Float16* __restrict__ attnA, const _Float16* __restrict__ woB,
                const float* __restrict__ bo, float* __restrict__ out) {
    const int wave = threadIdx.x >> 5;
    const int mtb  = (blockIdx.y * 4 + wave) * 2;
    const int nt0  = blockIdx.x * 4;
    const int lane = threadIdx.x & 31;

    v8f acc[2][4] = {};
#pragma unroll 2
    for (int kc = 0; kc < Dc / 32; ++kc) {
        const v16h a0 = *(const v16h*)(attnA + ((size_t)(mtb + 0) * 32 + kc) * TILE + lane * 16);
        const v16h a1 = *(const v16h*)(attnA + ((size_t)(mtb + 1) * 32 + kc) * TILE + lane * 16);
#pragma unroll
        for (int t = 0; t < 4; ++t) {
            const v16h b = *(const v16h*)(woB + ((size_t)kc * 64 + nt0 + t) * TILE + lane * 16);
            acc[0][t] = wmma_f16(a0, b, acc[0][t]);
            acc[1][t] = wmma_f16(a1, b, acc[1][t]);
        }
    }

    const int rowoff = (lane & 16) ? 8 : 0;
#pragma unroll
    for (int t = 0; t < 4; ++t) {
        const int   col = (nt0 + t) * 16 + (lane & 15);
        const float bsv = bo[col];
#pragma unroll
        for (int mi = 0; mi < 2; ++mi)
#pragma unroll
        for (int r = 0; r < 8; ++r) {
            const int m = (mtb + mi) * 16 + r + rowoff;
            out[(size_t)m * Dc + col] = acc[mi][t][r] + bsv;
        }
    }
}

// ---------------------------------------------------------------------------
extern "C" void kernel_launch(void* const* d_in, const int* in_sizes, int n_in,
                              void* d_out, int out_size, void* d_ws, size_t ws_size,
                              hipStream_t stream) {
    (void)in_sizes; (void)n_in; (void)out_size; (void)ws_size;

    const float*         x      = (const float*)d_in[0];
    const float*         memory = (const float*)d_in[1];
    const unsigned char* mask   = (const unsigned char*)d_in[2];  // jnp.bool_: 1B/elem
    const float* wq = (const float*)d_in[3];
    const float* bq = (const float*)d_in[4];
    const float* wk = (const float*)d_in[5];
    const float* bk = (const float*)d_in[6];
    const float* wv = (const float*)d_in[7];
    const float* bv = (const float*)d_in[8];
    const float* wo = (const float*)d_in[9];
    const float* bo = (const float*)d_in[10];
    float* out = (float*)d_out;

    const size_t AQ  = (size_t)Mc * Dc;                       // 4M halfs
    const size_t WB1 = (size_t)(Dc / 32) * (Dc / 16) * TILE;  // 1M halfs
    _Float16* xa    = (_Float16*)d_ws;
    _Float16* ma    = xa + AQ;
    _Float16* wB    = ma + AQ;
    _Float16* Qa    = wB + 4 * WB1;
    _Float16* Kb16  = Qa + AQ;
    _Float16* Vb    = Kb16 + AQ;
    _Float16* attnA = Vb + AQ;

    repack_a_kernel<<<dim3(8, Mc / 16), 128, 0, stream>>>(x, xa);
    repack_a_kernel<<<dim3(8, Mc / 16), 128, 0, stream>>>(memory, ma);
    repack_b_kernel<<<dim3(16, Dc / 32, 4), 128, 0, stream>>>(wq, wk, wv, wo, wB);

    qkv_proj_kernel<<<dim3(Dc / 64, Mc / 128, 3), 128, 0, stream>>>(
        xa, ma, wB, bq, bk, bv, Qa, Kb16, Vb);

    attn_kernel<<<dim3(Sc / 64, Hc, Bc), 128, 0, stream>>>(Qa, Kb16, Vb, mask, attnA);

    out_proj_kernel<<<dim3(Dc / 64, Mc / 128, 1), 128, 0, stream>>>(
        attnA, wB + 3 * WB1, bo, out);
}
